// T5_27462020891130
// MI455X (gfx1250) — compile-verified, matrix operand order checked by reading, not run on hardware
//
#include <hip/hip_runtime.h>
#include <math.h>

// ---------------------------------------------------------------------------
// Types
// ---------------------------------------------------------------------------
typedef __bf16 bf16;
typedef __attribute__((ext_vector_type(16))) __bf16 bf16x16;
typedef __attribute__((ext_vector_type(8)))  float  v8f;
typedef __attribute__((ext_vector_type(4)))  unsigned int u32x4;
typedef __attribute__((ext_vector_type(8)))  int i32x8;
typedef __attribute__((ext_vector_type(4)))  int i32x4;

struct alignas(16) f4 { float a, b, c, d; };
union Frag { bf16x16 v; f4 q[2]; };

// Problem constants
static constexpr int Bc = 4, Sc = 1024, SCc = 1024, Dc = 1024, Hc = 16, FFc = 4096;
static constexpr int HDc = Dc / Hc;              // 64

// GEMM tiling
static constexpr int BM = 128, BN = 128, BK = 32, BKP = 40; // BKP: padded LDS stride

// ---------------------------------------------------------------------------
// TDM 2-D tile load: 128 rows x 32 bf16, row stride `ld` elements in memory,
// destination LDS rows padded 64B data + 16B pad (matches [128][40] bf16).
// D# built per CDNA5 ISA 8.3/8.4: type=2, count=1, data_size=2B,
// pad_interval code 3 (16 DWORDs), pad_amount code 3 (4 DWORDs).
// ---------------------------------------------------------------------------
__device__ __forceinline__
void tdm_load_tile_128x32(unsigned ldsAddr, const bf16* gsrc, int ld)
{
    const unsigned long long ga = (unsigned long long)(size_t)gsrc;
    u32x4 g0;
    g0[0] = 1u;                                            // count=1 (valid)
    g0[1] = ldsAddr;                                       // lds_addr
    g0[2] = (unsigned)ga;                                  // global_addr[31:0]
    g0[3] = (unsigned)((ga >> 32) & 0x01FFFFFFu) | (2u << 30); // addr[56:32] | type=2
    i32x8 g1;
    g1[0] = (int)((1u << 16) | (1u << 20) | (3u << 22) | (3u << 25));
    //        data_size=2B    pad_enable   interval=16dw  amount=4dw
    g1[1] = (int)((unsigned)BK << 16);                     // tensor_dim0 lo16 = 32
    g1[2] = (int)((unsigned)BM << 16);                     // dim0 hi | tensor_dim1 lo = 128
    g1[3] = (int)((unsigned)BK << 16);                     // dim1 hi | tile_dim0 = 32
    g1[4] = BM;                                            // tile_dim1=128, tile_dim2=0
    g1[5] = ld;                                            // tensor_dim0_stride lo
    g1[6] = 0;
    g1[7] = 0;
    i32x4 gz = {0, 0, 0, 0};
#if defined(__clang_major__) && __clang_major__ >= 23
    i32x8 gz8 = {0, 0, 0, 0, 0, 0, 0, 0};
    __builtin_amdgcn_tensor_load_to_lds(g0, g1, gz, gz, gz8, 0);
#else
    __builtin_amdgcn_tensor_load_to_lds(g0, g1, gz, gz, 0);
#endif
}

// ---------------------------------------------------------------------------
// Shared epilogue helper
// ---------------------------------------------------------------------------
__device__ __forceinline__
void store_elem(bf16* Cb, float* Cf, bool outBf, const float* bias, int relMod,
                int r, int c, int ldc, float v)
{
    if (bias) v += bias[c];
    if (relMod) {
        int rel = c - (r % relMod);
        rel = rel < -3 ? -3 : (rel > 3 ? 3 : rel);
        v += (float)rel;
    }
    const long idx = (long)r * ldc + c;
    if (outBf) Cb[idx] = (bf16)v; else Cf[idx] = v;
}

// ---------------------------------------------------------------------------
// TDM-fed, double-buffered WMMA GEMM. Requires M%128==0, N%128==0, K%32==0.
// C[M,N] = A[M,K] * BT[N,K]^T (+bias) (+relclip). Two-level batch offsets.
// Wave 0 drives the Tensor Data Mover; all waves only do ds_load frags + WMMA.
// ---------------------------------------------------------------------------
__global__ __launch_bounds__(256)
void gemm_bf16_wmma_tdm(const bf16* __restrict__ A,  int lda, long sA1, long sA2, int dA,
                        const bf16* __restrict__ BT, int ldb, long sB1, long sB2, int dB,
                        void* __restrict__ Cv,       int ldc, long sC1, long sC2, int dC,
                        const float* __restrict__ bias,
                        int M, int N, int K, int flags, int relMod)
{
    const int z = blockIdx.z;
    A  += (long)(z / dA) * sA1 + (long)(z % dA) * sA2;
    BT += (long)(z / dB) * sB1 + (long)(z % dB) * sB2;
    const long cOff = (long)(z / dC) * sC1 + (long)(z % dC) * sC2;

    __shared__ bf16 lA[2][BM][BKP];
    __shared__ bf16 lB[2][BN][BKP];

    const int tid    = threadIdx.x;
    const int blockM = blockIdx.y * BM;
    const int blockN = blockIdx.x * BN;
    const int wave   = tid >> 5;
    const int lane   = tid & 31;
    const int hh     = lane >> 4;
    const int lrow   = lane & 15;
    const int waveM  = (wave >> 1) * 32;
    const int waveN  = (wave & 1) * 64;

    const bf16* baseA = A  + (long)blockM * lda;
    const bf16* baseB = BT + (long)blockN * ldb;
    const unsigned ldsA[2] = {(unsigned)(size_t)&lA[0][0][0], (unsigned)(size_t)&lA[1][0][0]};
    const unsigned ldsB[2] = {(unsigned)(size_t)&lB[0][0][0], (unsigned)(size_t)&lB[1][0][0]};

    v8f acc[2][4];
    #pragma unroll
    for (int mt = 0; mt < 2; ++mt)
        #pragma unroll
        for (int nt = 0; nt < 4; ++nt)
            acc[mt][nt] = (v8f){0.f,0.f,0.f,0.f,0.f,0.f,0.f,0.f};

    const int ktiles = K / BK;

    if (wave == 0) {                       // prime buffer 0 via TDM
        tdm_load_tile_128x32(ldsA[0], baseA, lda);
        tdm_load_tile_128x32(ldsB[0], baseB, ldb);
        __builtin_amdgcn_s_wait_tensorcnt(0);
    }
    __syncthreads();

    for (int kt = 0; kt < ktiles; ++kt) {
        const int buf = kt & 1;
        if (wave == 0 && kt + 1 < ktiles) {   // stream next tile into other buffer
            tdm_load_tile_128x32(ldsA[buf ^ 1], baseA + (long)(kt + 1) * BK, lda);
            tdm_load_tile_128x32(ldsB[buf ^ 1], baseB + (long)(kt + 1) * BK, ldb);
        }

        const int kb = hh * 8;             // ISA bf16 A-frag K layout
        Frag af[2], bfr[4];
        #pragma unroll
        for (int mt = 0; mt < 2; ++mt) {
            const int r = waveM + mt * 16 + lrow;
            af[mt].q[0] = *reinterpret_cast<const f4*>(&lA[buf][r][kb]);
            af[mt].q[1] = *reinterpret_cast<const f4*>(&lA[buf][r][kb + 16]);
        }
        #pragma unroll
        for (int nt = 0; nt < 4; ++nt) {
            const int r = waveN + nt * 16 + lrow;
            bfr[nt].q[0] = *reinterpret_cast<const f4*>(&lB[buf][r][kb]);
            bfr[nt].q[1] = *reinterpret_cast<const f4*>(&lB[buf][r][kb + 16]);
        }
        #pragma unroll
        for (int mt = 0; mt < 2; ++mt)
            #pragma unroll
            for (int nt = 0; nt < 4; ++nt)
                acc[mt][nt] = __builtin_amdgcn_wmma_f32_16x16x32_bf16(
                    false, af[mt].v, false, bfr[nt].v,
                    (short)0, acc[mt][nt], false, false);

        if (wave == 0 && kt + 1 < ktiles)
            __builtin_amdgcn_s_wait_tensorcnt(0);
        __syncthreads();                   // single barrier per k-step
    }

    const bool outBf = (flags & 1) != 0;
    bf16*  Cb = reinterpret_cast<bf16*>(Cv) + cOff;
    float* Cf = reinterpret_cast<float*>(Cv) + cOff;
    #pragma unroll
    for (int mt = 0; mt < 2; ++mt)
        #pragma unroll
        for (int nt = 0; nt < 4; ++nt) {
            const int c = blockN + waveN + nt * 16 + lrow;
            #pragma unroll
            for (int g = 0; g < 8; ++g) {
                const int r = blockM + waveM + mt * 16 + hh * 8 + g;
                store_elem(Cb, Cf, outBf, bias, relMod, r, c, ldc, acc[mt][nt][g]);
            }
        }
}

// ---------------------------------------------------------------------------
// Classic guarded WMMA GEMM (for N or M not multiples of the tile, e.g. AV
// with N=64). Clamped row index -> unconditional global_load_b128 + cndmask.
// ---------------------------------------------------------------------------
__global__ __launch_bounds__(256)
void gemm_bf16_wmma_guarded(const bf16* __restrict__ A,  int lda, long sA1, long sA2, int dA,
                            const bf16* __restrict__ BT, int ldb, long sB1, long sB2, int dB,
                            void* __restrict__ Cv,       int ldc, long sC1, long sC2, int dC,
                            const float* __restrict__ bias,
                            int M, int N, int K, int flags, int relMod)
{
    const int z = blockIdx.z;
    A  += (long)(z / dA) * sA1 + (long)(z % dA) * sA2;
    BT += (long)(z / dB) * sB1 + (long)(z % dB) * sB2;
    const long cOff = (long)(z / dC) * sC1 + (long)(z % dC) * sC2;

    __shared__ bf16 lA[BM][BKP];
    __shared__ bf16 lB[BN][BKP];

    const int tid   = threadIdx.x;
    const int row2  = tid >> 1;
    const int colh  = (tid & 1) * 16;
    const int blockM = blockIdx.y * BM;
    const int blockN = blockIdx.x * BN;
    const int wave  = tid >> 5;
    const int lane  = tid & 31;
    const int hh    = lane >> 4;
    const int lrow  = lane & 15;
    const int waveM = (wave >> 1) * 32;
    const int waveN = (wave & 1) * 64;

    v8f acc[2][4];
    #pragma unroll
    for (int mt = 0; mt < 2; ++mt)
        #pragma unroll
        for (int nt = 0; nt < 4; ++nt)
            acc[mt][nt] = (v8f){0.f,0.f,0.f,0.f,0.f,0.f,0.f,0.f};

    const int ktiles = K / BK;
    const f4 z4 = {0.f, 0.f, 0.f, 0.f};
    const int ar = blockM + row2;
    const int br = blockN + row2;
    const bool aOk = ar < M, bOk = br < N;
    const bf16* pA = A  + (long)(aOk ? ar : (M - 1)) * lda + colh;  // clamped: always valid
    const bf16* pB = BT + (long)(bOk ? br : (N - 1)) * ldb + colh;
    f4 ra0, ra1, rb0, rb1;

    auto loadTile = [&](int kt) {
        const long kcol = (long)kt * BK;
        ra0 = *reinterpret_cast<const f4*>(pA + kcol);
        ra1 = *reinterpret_cast<const f4*>(pA + kcol + 8);
        rb0 = *reinterpret_cast<const f4*>(pB + kcol);
        rb1 = *reinterpret_cast<const f4*>(pB + kcol + 8);
        if (!aOk) { ra0 = z4; ra1 = z4; }
        if (!bOk) { rb0 = z4; rb1 = z4; }
    };

    loadTile(0);
    for (int kt = 0; kt < ktiles; ++kt) {
        __syncthreads();
        *reinterpret_cast<f4*>(&lA[row2][colh])     = ra0;
        *reinterpret_cast<f4*>(&lA[row2][colh + 8]) = ra1;
        *reinterpret_cast<f4*>(&lB[row2][colh])     = rb0;
        *reinterpret_cast<f4*>(&lB[row2][colh + 8]) = rb1;
        __syncthreads();
        if (kt + 1 < ktiles) loadTile(kt + 1);

        const int kb = hh * 8;
        Frag af[2], bfr[4];
        #pragma unroll
        for (int mt = 0; mt < 2; ++mt) {
            const int r = waveM + mt * 16 + lrow;
            af[mt].q[0] = *reinterpret_cast<const f4*>(&lA[r][kb]);
            af[mt].q[1] = *reinterpret_cast<const f4*>(&lA[r][kb + 16]);
        }
        #pragma unroll
        for (int nt = 0; nt < 4; ++nt) {
            const int r = waveN + nt * 16 + lrow;
            bfr[nt].q[0] = *reinterpret_cast<const f4*>(&lB[r][kb]);
            bfr[nt].q[1] = *reinterpret_cast<const f4*>(&lB[r][kb + 16]);
        }
        #pragma unroll
        for (int mt = 0; mt < 2; ++mt)
            #pragma unroll
            for (int nt = 0; nt < 4; ++nt)
                acc[mt][nt] = __builtin_amdgcn_wmma_f32_16x16x32_bf16(
                    false, af[mt].v, false, bfr[nt].v,
                    (short)0, acc[mt][nt], false, false);
    }

    const bool outBf = (flags & 1) != 0;
    bf16*  Cb = reinterpret_cast<bf16*>(Cv) + cOff;
    float* Cf = reinterpret_cast<float*>(Cv) + cOff;
    #pragma unroll
    for (int mt = 0; mt < 2; ++mt)
        #pragma unroll
        for (int nt = 0; nt < 4; ++nt) {
            const int c = blockN + waveN + nt * 16 + lrow;
            #pragma unroll
            for (int g = 0; g < 8; ++g) {
                const int r = blockM + waveM + mt * 16 + hh * 8 + g;
                if (r < M && c < N)
                    store_elem(Cb, Cf, outBf, bias, relMod, r, c, ldc, acc[mt][nt][g]);
            }
        }
}

// ---------------------------------------------------------------------------
// Masked, scaled softmax (matches reference: score*scale*mask, softmax all).
// ---------------------------------------------------------------------------
__global__ __launch_bounds__(256)
void softmax_mask(const float* __restrict__ scores, bf16* __restrict__ probs,
                  int Sn, int SCn, float scale)
{
    const long row = blockIdx.x;
    const int  i   = (int)(row % Sn);
    const float* src = scores + row * SCn;
    bf16*        dst = probs  + row * SCn;
    __shared__ float red[256];
    const int tid = threadIdx.x;

    float mx = -3.0e38f;
    for (int j = tid; j < SCn; j += 256) {
        const float m = (i >= j - SCn + Sn) ? 1.f : 0.f;
        mx = fmaxf(mx, src[j] * scale * m);
    }
    red[tid] = mx; __syncthreads();
    for (int s2 = 128; s2 > 0; s2 >>= 1) {
        if (tid < s2) red[tid] = fmaxf(red[tid], red[tid + s2]);
        __syncthreads();
    }
    mx = red[0]; __syncthreads();

    float sum = 0.f;
    for (int j = tid; j < SCn; j += 256) {
        const float m = (i >= j - SCn + Sn) ? 1.f : 0.f;
        sum += __expf(src[j] * scale * m - mx);
    }
    red[tid] = sum; __syncthreads();
    for (int s2 = 128; s2 > 0; s2 >>= 1) {
        if (tid < s2) red[tid] += red[tid + s2];
        __syncthreads();
    }
    const float inv = 1.f / red[0];

    for (int j = tid; j < SCn; j += 256) {
        const float m = (i >= j - SCn + Sn) ? 1.f : 0.f;
        dst[j] = (bf16)(__expf(src[j] * scale * m - mx) * inv);
    }
}

// ---------------------------------------------------------------------------
// Fused residual add + LayerNorm; writes f32 and optional bf16 copy.
// ---------------------------------------------------------------------------
__global__ __launch_bounds__(256)
void add_ln(const float* __restrict__ x, const float* __restrict__ y,
            const float* __restrict__ gamma, const float* __restrict__ beta,
            float* __restrict__ outF, bf16* __restrict__ outB, int Dm)
{
    const long row = blockIdx.x;
    const float* px = x + row * Dm;
    const float* py = y + row * Dm;
    __shared__ float red[256];
    const int tid = threadIdx.x;

    float vv[4];
    float s = 0.f;
    int u = 0;
    for (int j = tid; j < Dm; j += 256, ++u) { vv[u] = px[j] + py[j]; s += vv[u]; }
    red[tid] = s; __syncthreads();
    for (int s2 = 128; s2 > 0; s2 >>= 1) { if (tid < s2) red[tid] += red[tid + s2]; __syncthreads(); }
    const float mu = red[0] / (float)Dm; __syncthreads();

    float q = 0.f; u = 0;
    for (int j = tid; j < Dm; j += 256, ++u) { const float d = vv[u] - mu; q += d * d; }
    red[tid] = q; __syncthreads();
    for (int s2 = 128; s2 > 0; s2 >>= 1) { if (tid < s2) red[tid] += red[tid + s2]; __syncthreads(); }
    const float rstd = rsqrtf(red[0] / (float)Dm + 1e-6f);

    u = 0;
    for (int j = tid; j < Dm; j += 256, ++u) {
        const float t = (vv[u] - mu) * rstd * gamma[j] + beta[j];
        outF[row * Dm + j] = t;
        if (outB) outB[row * Dm + j] = (bf16)t;
    }
}

// ---------------------------------------------------------------------------
// GEGLU
// ---------------------------------------------------------------------------
__global__ __launch_bounds__(256)
void geglu_kernel(const float* __restrict__ t2, bf16* __restrict__ g,
                  long total, int FFn)
{
    const long idx = (long)blockIdx.x * 256 + threadIdx.x;
    if (idx >= total) return;
    const long r = idx / FFn;
    const int  c = (int)(idx % FFn);
    const float a  = t2[r * (2 * FFn) + c];
    const float gt = t2[r * (2 * FFn) + FFn + c];
    const float th = tanhf(gt * 0.7978845608f * (1.f + 0.044715f * gt * gt));
    g[idx] = (bf16)(a * 0.5f * gt * (1.f + th));
}

// ---------------------------------------------------------------------------
// Elementwise f32 -> bf16
// ---------------------------------------------------------------------------
__global__ __launch_bounds__(256)
void f32_to_bf16(const float* __restrict__ in, bf16* __restrict__ out, long n)
{
    const long idx = (long)blockIdx.x * 256 + threadIdx.x;
    if (idx < n) out[idx] = (bf16)in[idx];
}

// ---------------------------------------------------------------------------
// Tiled transpose f32 [rows,cols] -> bf16 [cols,rows]
// ---------------------------------------------------------------------------
__global__ __launch_bounds__(256)
void transpose_f32_bf16(const float* __restrict__ in, bf16* __restrict__ out,
                        int rows, int cols)
{
    __shared__ float tile[32][33];
    const int c0 = blockIdx.x * 32, r0 = blockIdx.y * 32;
    const int tx = threadIdx.x, ty = threadIdx.y;
    for (int u = 0; u < 32; u += 8) {
        const int r = r0 + ty + u, c = c0 + tx;
        tile[ty + u][tx] = (r < rows && c < cols) ? in[(long)r * cols + c] : 0.f;
    }
    __syncthreads();
    for (int u = 0; u < 32; u += 8) {
        const int c = c0 + ty + u, r = r0 + tx;
        if (c < cols && r < rows) out[(long)c * rows + r] = (bf16)tile[tx][ty + u];
    }
}

// ---------------------------------------------------------------------------
// Per-(b,h) transpose of V: v[b, i, h*hd+d] -> vT[(b*H+h), d, i]   (bf16)
// ---------------------------------------------------------------------------
__global__ __launch_bounds__(256)
void transpose_heads_bf16(const bf16* __restrict__ v, bf16* __restrict__ vT,
                          int Skv, int Dm, int hd, int Hn)
{
    const int z = blockIdx.z, b = z / Hn, h = z % Hn;
    const bf16* src = v  + (long)b * Skv * Dm + h * hd;
    bf16*       dst = vT + (long)z * hd * Skv;
    __shared__ bf16 tile[32][33];
    const int i0 = blockIdx.x * 32, d0 = blockIdx.y * 32;
    const int tx = threadIdx.x, ty = threadIdx.y;
    for (int u = 0; u < 32; u += 8) {
        const int i = i0 + ty + u, d = d0 + tx;
        tile[ty + u][tx] = (i < Skv && d < hd) ? src[(long)i * Dm + d] : (bf16)0.f;
    }
    __syncthreads();
    for (int u = 0; u < 32; u += 8) {
        const int d = d0 + ty + u, i = i0 + tx;
        if (d < hd && i < Skv) dst[(long)d * Skv + i] = tile[tx][ty + u];
    }
}

// ---------------------------------------------------------------------------
// Host orchestration
// ---------------------------------------------------------------------------
extern "C" void kernel_launch(void* const* d_in, const int* in_sizes, int n_in,
                              void* d_out, int out_size, void* d_ws, size_t ws_size,
                              hipStream_t stream)
{
    (void)in_sizes; (void)n_in; (void)out_size; (void)ws_size;
    const int B = Bc, S = Sc, SC = SCc, D = Dc, H = Hc, FF = FFc, HD = HDc;

    const float* x   = (const float*)d_in[0];
    const float* ctx = (const float*)d_in[1];
    const float* Wq[2] = {(const float*)d_in[2],  (const float*)d_in[10]};
    const float* bq[2] = {(const float*)d_in[3],  (const float*)d_in[11]};
    const float* Wk[2] = {(const float*)d_in[4],  (const float*)d_in[12]};
    const float* bk[2] = {(const float*)d_in[5],  (const float*)d_in[13]};
    const float* Wv[2] = {(const float*)d_in[6],  (const float*)d_in[14]};
    const float* bv[2] = {(const float*)d_in[7],  (const float*)d_in[15]};
    const float* Wo[2] = {(const float*)d_in[8],  (const float*)d_in[16]};
    const float* bo[2] = {(const float*)d_in[9],  (const float*)d_in[17]};
    const float* W1 = (const float*)d_in[18]; const float* b1 = (const float*)d_in[19];
    const float* Wg = (const float*)d_in[20]; const float* bg = (const float*)d_in[21];
    const float* W2 = (const float*)d_in[22]; const float* b2 = (const float*)d_in[23];
    const float* lns[3] = {(const float*)d_in[24], (const float*)d_in[26], (const float*)d_in[28]};
    const float* lnb[3] = {(const float*)d_in[25], (const float*)d_in[27], (const float*)d_in[29]};

    // ---- workspace carve-up ----
    char* W = (char*)d_ws;
    size_t off = 0;
    auto take = [&](size_t bytes) -> void* {
        void* p = W + off; off += (bytes + 255) & ~(size_t)255; return p;
    };
    bf16* wTq[2]; bf16* wTk[2]; bf16* wTv[2]; bf16* wTo[2];
    for (int l = 0; l < 2; ++l) {
        wTq[l] = (bf16*)take((size_t)D * D * 2);
        wTk[l] = (bf16*)take((size_t)D * D * 2);
        wTv[l] = (bf16*)take((size_t)D * D * 2);
        wTo[l] = (bf16*)take((size_t)D * D * 2);
    }
    bf16* W1T = (bf16*)take((size_t)D * FF * 2);
    bf16* WgT = (bf16*)take((size_t)FF * 2 * FF * 2);
    bf16* W2T = (bf16*)take((size_t)FF * D * 2);
    bf16* xbf   = (bf16*)take((size_t)B * S  * D * 2);
    bf16* ctxbf = (bf16*)take((size_t)B * SC * D * 2);
    bf16* qbf = (bf16*)take((size_t)B * S  * D * 2);
    bf16* kbf = (bf16*)take((size_t)B * SC * D * 2);
    bf16* vbf = (bf16*)take((size_t)B * SC * D * 2);
    bf16* vTb = (bf16*)take((size_t)B * H * HD * SC * 2);
    float* scoresF = (float*)take((size_t)B * H * S * SC * 4);   // reused as t2 (B*S*2FF f32)
    bf16*  probsB  = (bf16*)take((size_t)B * H * S * SC * 2);    // reused as geglu out
    bf16*  avB   = (bf16*)take((size_t)B * S * D * 2);
    float* attnF = (float*)take((size_t)B * S * D * 4);
    float* h1    = (float*)take((size_t)B * S * D * 4);
    bf16*  h1bf  = (bf16*)take((size_t)B * S * D * 2);
    float* h2    = (float*)take((size_t)B * S * D * 4);
    bf16*  h2bf  = (bf16*)take((size_t)B * S * D * 2);
    bf16*  t1bf  = (bf16*)take((size_t)B * S * FF * 2);
    float* t3f   = (float*)take((size_t)B * S * D * 4);
    float* t2f   = scoresF;
    bf16*  gbf   = probsB;

    const dim3 tb(32, 8);
    auto gemm = [&](const bf16* A, int lda, long sA1, long sA2, int dA,
                    const bf16* BT, int ldb, long sB1, long sB2, int dB,
                    void* C, int ldc, long sC1, long sC2, int dC,
                    const float* bias, int M, int N, int K, int Z,
                    bool outBf, int relMod) {
        dim3 grid((unsigned)((N + BN - 1) / BN), (unsigned)((M + BM - 1) / BM), (unsigned)Z);
        if ((M % BM) == 0 && (N % BN) == 0)
            gemm_bf16_wmma_tdm<<<grid, 256, 0, stream>>>(A, lda, sA1, sA2, dA,
                                                         BT, ldb, sB1, sB2, dB,
                                                         C, ldc, sC1, sC2, dC,
                                                         bias, M, N, K, outBf ? 1 : 0, relMod);
        else
            gemm_bf16_wmma_guarded<<<grid, 256, 0, stream>>>(A, lda, sA1, sA2, dA,
                                                             BT, ldb, sB1, sB2, dB,
                                                             C, ldc, sC1, sC2, dC,
                                                             bias, M, N, K, outBf ? 1 : 0, relMod);
    };

    // ---- weight transpose-convert (f32 [K,N] -> bf16 [N,K]) ----
    for (int l = 0; l < 2; ++l) {
        transpose_f32_bf16<<<dim3(D / 32, D / 32), tb, 0, stream>>>(Wq[l], wTq[l], D, D);
        transpose_f32_bf16<<<dim3(D / 32, D / 32), tb, 0, stream>>>(Wk[l], wTk[l], D, D);
        transpose_f32_bf16<<<dim3(D / 32, D / 32), tb, 0, stream>>>(Wv[l], wTv[l], D, D);
        transpose_f32_bf16<<<dim3(D / 32, D / 32), tb, 0, stream>>>(Wo[l], wTo[l], D, D);
    }
    transpose_f32_bf16<<<dim3(FF / 32, D / 32), tb, 0, stream>>>(W1, W1T, D, FF);
    transpose_f32_bf16<<<dim3(2 * FF / 32, FF / 32), tb, 0, stream>>>(Wg, WgT, FF, 2 * FF);
    transpose_f32_bf16<<<dim3(D / 32, FF / 32), tb, 0, stream>>>(W2, W2T, FF, D);

    f32_to_bf16<<<(B * S * D + 255) / 256, 256, 0, stream>>>(x, xbf, (long)B * S * D);
    f32_to_bf16<<<(B * SC * D + 255) / 256, 256, 0, stream>>>(ctx, ctxbf, (long)B * SC * D);

    const float scale = 1.0f / 32.0f;  // 1/sqrt(D), D=1024

    auto attention = [&](const bf16* qin, const bf16* kvin, int Skv, int l,
                         const float* resid, const float* g, const float* bta,
                         float* hOut, bf16* hbfOut) {
        gemm(qin,  D, 0, 0, 1, wTq[l], D, 0, 0, 1, qbf, D, 0, 0, 1,
             bq[l], B * S, D, D, 1, true, S);
        gemm(kvin, D, 0, 0, 1, wTk[l], D, 0, 0, 1, kbf, D, 0, 0, 1,
             bk[l], B * Skv, D, D, 1, true, 0);
        gemm(kvin, D, 0, 0, 1, wTv[l], D, 0, 0, 1, vbf, D, 0, 0, 1,
             bv[l], B * Skv, D, D, 1, true, Skv);
        transpose_heads_bf16<<<dim3(Skv / 32, HD / 32, B * H), tb, 0, stream>>>(
            vbf, vTb, Skv, D, HD, H);
        // scores[z] = qh @ kh^T  (z = b*H + h)
        gemm(qbf, D, (long)S * D, HD, H,
             kbf, D, (long)Skv * D, HD, H,
             scoresF, Skv, (long)S * Skv, 0, 1,
             nullptr, S, Skv, HD, B * H, false, 0);
        softmax_mask<<<B * H * S, 256, 0, stream>>>(scoresF, probsB, S, Skv, scale);
        // av[z] = probs @ v  via vT  (N=64 -> guarded kernel)
        gemm(probsB, Skv, (long)S * Skv, 0, 1,
             vTb, Skv, (long)HD * Skv, 0, 1,
             avB, D, (long)S * D, HD, H,
             nullptr, S, HD, Skv, B * H, true, 0);
        gemm(avB, D, 0, 0, 1, wTo[l], D, 0, 0, 1, attnF, D, 0, 0, 1,
             bo[l], B * S, D, D, 1, false, 0);
        add_ln<<<B * S, 256, 0, stream>>>(resid, attnF, g, bta, hOut, hbfOut, D);
    };

    attention(xbf, xbf, S, 0, x, lns[0], lnb[0], h1, h1bf);
    attention(h1bf, ctxbf, SC, 1, h1, lns[1], lnb[1], h2, h2bf);

    // FFN
    gemm(h2bf, D, 0, 0, 1, W1T, D, 0, 0, 1, t1bf, FF, 0, 0, 1,
         b1, B * S, FF, D, 1, true, 0);
    gemm(t1bf, FF, 0, 0, 1, WgT, FF, 0, 0, 1, t2f, 2 * FF, 0, 0, 1,
         bg, B * S, 2 * FF, FF, 1, false, 0);
    geglu_kernel<<<(B * S * FF + 255) / 256, 256, 0, stream>>>(
        t2f, gbf, (long)B * S * FF, FF);
    gemm(gbf, FF, 0, 0, 1, W2T, FF, 0, 0, 1, t3f, D, 0, 0, 1,
         b2, B * S, D, FF, 1, false, 0);
    add_ln<<<B * S, 256, 0, stream>>>(h2, t3f, lns[2], lnb[2],
                                      (float*)d_out, (bf16*)nullptr, D);
}
